// SDCN_89558658056651
// MI455X (gfx1250) — compile-verified, matrix operand order checked by reading, not run on hardware
//
#include <hip/hip_runtime.h>
#include <hip/hip_bf16.h>
#include <cstdint>
#include <cstddef>

// ---------------------------------------------------------------------------
// Types
// ---------------------------------------------------------------------------
typedef __bf16 bhalf;
typedef __attribute__((ext_vector_type(16))) __bf16 bf16x16;
typedef __attribute__((ext_vector_type(8)))  __bf16 bf16x8;
typedef __attribute__((ext_vector_type(8)))  float  f32x8;

__device__ __forceinline__ bhalf f2bf(float x) {
    union { float f; unsigned u; } v; v.f = x;
    unsigned r = v.u + 0x7FFFu + ((v.u >> 16) & 1u);   // round-to-nearest-even
    union { unsigned short s; bhalf b; } o;
    o.s = (unsigned short)(r >> 16);
    return o.b;
}

// ---------------------------------------------------------------------------
// WMMA GEMM: C[Mp,Np] = A[Mp,Kp](bf16) * Bt[Np,Kp]^T(bf16) + bias
// All dims multiples of 64, zero padded -> no bounds checks.
// 4 waves/block; block tile 64x64; wave tile 32x32 (2x2 wmma frags).
// K-step 64: 64x64 bf16 A/B tiles staged to LDS via
// GLOBAL_LOAD_ASYNC_TO_LDS_B128 (ASYNCcnt), double buffered (32KB LDS);
// one s_wait_asynccnt + one barrier per 8 WMMAs.
// ---------------------------------------------------------------------------
__global__ __launch_bounds__(128)
void sdcn_wmma_gemm_bf16(const bhalf* __restrict__ A,
                         const bhalf* __restrict__ Bt,
                         const float* __restrict__ bias,   // may be null
                         float* __restrict__ C,
                         int Mp, int Kp, int Np, int Nreal)
{
    // double-buffered 64x64 bf16 tiles: 8KB per tile, 32KB total
    __shared__ bhalf sA[2][64 * 64];
    __shared__ bhalf sB[2][64 * 64];

    const int tid  = threadIdx.x;
    const int lane = tid & 31;
    const int wave = tid >> 5;             // 0..3
    const int wm   = wave >> 1;            // 0..1 (M half of block tile)
    const int wn   = wave & 1;             // 0..1 (N half of block tile)
    const int bm0  = blockIdx.y * 64;
    const int bn0  = blockIdx.x * 64;
    const int sub  = lane & 15;
    const int hi   = lane >> 4;

    // 32-bit LDS offsets (low 32 bits of flat shared address = LDS offset)
    const unsigned ldsA0 = (unsigned)(size_t)&sA[0][0];
    const unsigned ldsB0 = (unsigned)(size_t)&sB[0][0];

    // Stage a 64x64 bf16 tile (512 x 16B chunks) for K-window `k` into
    // buffer `buf`. Each of 128 threads issues 4 A-chunks + 4 B-chunks,
    // straight-line (constant trip count -> no exec masking).
    auto issue_tile = [&](int buf, int k) {
        #pragma unroll
        for (int i = 0; i < 4; ++i) {
            const int c    = tid + (i << 7);           // 0..511
            const int row  = c >> 3;
            const int offe = (c & 7) << 3;             // element offset in row
            const bhalf* gp = A + (size_t)(bm0 + row) * Kp + k + offe;
            unsigned la = ldsA0 + (unsigned)(buf << 13) + (unsigned)(c << 4);
            asm volatile("global_load_async_to_lds_b128 %0, %1, off"
                         :: "v"(la), "v"((unsigned long long)(size_t)gp)
                         : "memory");
        }
        #pragma unroll
        for (int i = 0; i < 4; ++i) {
            const int c    = tid + (i << 7);
            const int row  = c >> 3;
            const int offe = (c & 7) << 3;
            const bhalf* gp = Bt + (size_t)(bn0 + row) * Kp + k + offe;
            unsigned la = ldsB0 + (unsigned)(buf << 13) + (unsigned)(c << 4);
            asm volatile("global_load_async_to_lds_b128 %0, %1, off"
                         :: "v"(la), "v"((unsigned long long)(size_t)gp)
                         : "memory");
        }
    };

    f32x8 z8 = {0.f,0.f,0.f,0.f,0.f,0.f,0.f,0.f};
    f32x8 acc00 = z8, acc01 = z8, acc10 = z8, acc11 = z8;

    union Frag { bf16x16 v; bf16x8 h[2]; };

    issue_tile(0, 0);

    int buf = 0;
    for (int k = 0; k < Kp; k += 64) {
        // complete this wave's DMA for tile `buf`, then publish across waves
        asm volatile("s_wait_asynccnt 0x0" ::: "memory");
        __syncthreads();

        if (k + 64 < Kp) issue_tile(buf ^ 1, k + 64);      // overlap next DMA

        const bhalf* aBase = &sA[buf][(wm * 32 + sub) * 64];
        const bhalf* bBase = &sB[buf][(wn * 32 + sub) * 64];

        #pragma unroll
        for (int kc = 0; kc < 2; ++kc) {
            const int kk = kc * 32 + hi * 8;
            const bhalf* aR0 = aBase + kk;
            const bhalf* aR1 = aR0 + 16 * 64;
            const bhalf* bR0 = bBase + kk;
            const bhalf* bR1 = bR0 + 16 * 64;

            Frag a0, a1, b0, b1;
            a0.h[0] = *(const bf16x8*)(aR0);
            a0.h[1] = *(const bf16x8*)(aR0 + 16);
            a1.h[0] = *(const bf16x8*)(aR1);
            a1.h[1] = *(const bf16x8*)(aR1 + 16);
            b0.h[0] = *(const bf16x8*)(bR0);
            b0.h[1] = *(const bf16x8*)(bR0 + 16);
            b1.h[0] = *(const bf16x8*)(bR1);
            b1.h[1] = *(const bf16x8*)(bR1 + 16);

            acc00 = __builtin_amdgcn_wmma_f32_16x16x32_bf16(false, a0.v, false, b0.v, (short)0, acc00, false, false);
            acc01 = __builtin_amdgcn_wmma_f32_16x16x32_bf16(false, a0.v, false, b1.v, (short)0, acc01, false, false);
            acc10 = __builtin_amdgcn_wmma_f32_16x16x32_bf16(false, a1.v, false, b0.v, (short)0, acc10, false, false);
            acc11 = __builtin_amdgcn_wmma_f32_16x16x32_bf16(false, a1.v, false, b1.v, (short)0, acc11, false, false);
        }

        buf ^= 1;
    }

    // C/D layout: lane 0-15 col=lane rows 0..7; lane 16-31 col=lane-16 rows 8..15
    f32x8 accs[2][2] = {{acc00, acc01}, {acc10, acc11}};
    const int bm = bm0 + wm * 32;
    const int bn = bn0 + wn * 32;
    #pragma unroll
    for (int mi = 0; mi < 2; ++mi) {
        #pragma unroll
        for (int ni = 0; ni < 2; ++ni) {
            const int col  = bn + ni * 16 + sub;
            const int row0 = bm + mi * 16 + hi * 8;
            const float bz = (bias != nullptr && col < Nreal) ? bias[col] : 0.0f;
            float* cp = C + (size_t)row0 * Np + col;
            #pragma unroll
            for (int r = 0; r < 8; ++r)
                cp[(size_t)r * Np] = accs[mi][ni][r] + bz;
        }
    }
}

// ---------------------------------------------------------------------------
// Prep / elementwise kernels
// ---------------------------------------------------------------------------
__global__ void sdcn_transpose_cast(const float* __restrict__ W, // [K,Nt] row-major
                                    bhalf* __restrict__ Wt,      // [Np,Kp]
                                    int K, int Nt, int Kp, int Np)
{
    size_t idx = (size_t)blockIdx.x * blockDim.x + threadIdx.x;
    size_t total = (size_t)Np * Kp;
    if (idx >= total) return;
    int n = (int)(idx / Kp);
    int k = (int)(idx % Kp);
    float v = (n < Nt && k < K) ? W[(size_t)k * Nt + n] : 0.0f;
    Wt[idx] = f2bf(v);
}

__global__ void sdcn_cast_pad(const float* __restrict__ X, bhalf* __restrict__ Xb,
                              int M, int K, int Mp, int Kp)
{
    size_t idx = (size_t)blockIdx.x * blockDim.x + threadIdx.x;
    size_t total = (size_t)Mp * Kp;
    if (idx >= total) return;
    int r = (int)(idx / Kp);
    int k = (int)(idx % Kp);
    float v = (r < M && k < K) ? X[(size_t)r * K + k] : 0.0f;
    Xb[idx] = f2bf(v);
}

__global__ void sdcn_cast_guard(const float* __restrict__ S, bhalf* __restrict__ D,
                                int N, int Nreal, int Mp, int Np)
{
    size_t idx = (size_t)blockIdx.x * blockDim.x + threadIdx.x;
    size_t total = (size_t)Mp * Np;
    if (idx >= total) return;
    int r = (int)(idx / Np);
    int j = (int)(idx % Np);
    float v = (r < N && j < Nreal) ? S[idx] : 0.0f;
    D[idx] = f2bf(v);
}

__global__ void sdcn_zero(float* __restrict__ p, size_t n)
{
    size_t i = (size_t)blockIdx.x * blockDim.x + threadIdx.x;
    if (i < n) p[i] = 0.0f;
}

__global__ void sdcn_colstats(const float* __restrict__ C, int Nrows, int Np,
                              float* __restrict__ mean, float* __restrict__ rstd)
{
    const int j = blockIdx.x;                // column < Nreal
    float s = 0.f, s2 = 0.f;
    for (int r = threadIdx.x; r < Nrows; r += blockDim.x) {
        float v = C[(size_t)r * Np + j];
        s += v; s2 += v * v;
    }
    __shared__ float sh[256], sh2[256];
    sh[threadIdx.x] = s; sh2[threadIdx.x] = s2;
    __syncthreads();
    for (int o = 128; o > 0; o >>= 1) {
        if ((int)threadIdx.x < o) { sh[threadIdx.x] += sh[threadIdx.x + o]; sh2[threadIdx.x] += sh2[threadIdx.x + o]; }
        __syncthreads();
    }
    if (threadIdx.x == 0) {
        float m = sh[0] / (float)Nrows;
        float var = sh2[0] / (float)Nrows - m * m;
        mean[j] = m;
        rstd[j] = rsqrtf(fmaxf(var, 0.0f) + 1e-5f);
    }
}

__global__ void sdcn_bn_relu(float* __restrict__ C, bhalf* __restrict__ Cb,
                             const float* __restrict__ mean, const float* __restrict__ rstd,
                             const float* __restrict__ g, const float* __restrict__ b,
                             int N, int Nreal, int Mp, int Np)
{
    size_t idx = (size_t)blockIdx.x * blockDim.x + threadIdx.x;
    size_t total = (size_t)Mp * Np;
    if (idx >= total) return;
    int r = (int)(idx / Np);
    int j = (int)(idx % Np);
    float y = 0.0f;
    if (r < N && j < Nreal) {
        float v = C[idx];
        y = fmaxf((v - mean[j]) * rstd[j] * g[j] + b[j], 0.0f);
    }
    C[idx]  = y;
    Cb[idx] = f2bf(y);
}

__global__ void sdcn_combine_cast(const float* __restrict__ G, const float* __restrict__ H,
                                  bhalf* __restrict__ D, float sigma,
                                  int N, int Nreal, int Mp, int Np)
{
    size_t idx = (size_t)blockIdx.x * blockDim.x + threadIdx.x;
    size_t total = (size_t)Mp * Np;
    if (idx >= total) return;
    int r = (int)(idx / Np);
    int j = (int)(idx % Np);
    float v = 0.0f;
    if (r < N && j < Nreal) v = (1.0f - sigma) * G[idx] + sigma * H[idx];
    D[idx] = f2bf(v);
}

__global__ void sdcn_scatter(const float* __restrict__ Sup, float* __restrict__ Out,
                             const int* __restrict__ src, const int* __restrict__ dst,
                             const float* __restrict__ w, int dout, int ldc)
{
    const int e = blockIdx.x;
    const int j = blockIdx.y * blockDim.x + threadIdx.x;
    if (j >= dout) return;
    const int s = src[e];
    const int d = dst[e];
    const float wv = w[e];
    atomicAdd(&Out[(size_t)d * ldc + j], wv * Sup[(size_t)s * ldc + j]);
}

__global__ void sdcn_bias_add(float* __restrict__ Out, const float* __restrict__ b,
                              int N, int dout, int ldc)
{
    size_t idx = (size_t)blockIdx.x * blockDim.x + threadIdx.x;
    size_t total = (size_t)N * dout;
    if (idx >= total) return;
    int r = (int)(idx / dout);
    int j = (int)(idx % dout);
    Out[(size_t)r * ldc + j] += b[j];
}

__global__ void sdcn_softmax_h(const float* __restrict__ HG, int N, int ldh,
                               float* __restrict__ out_h, float* __restrict__ out_pred)
{
    int r = blockIdx.x * blockDim.x + threadIdx.x;
    if (r >= N) return;
    float v[10];
    float mx = -1e30f;
    #pragma unroll
    for (int k = 0; k < 10; ++k) { v[k] = HG[(size_t)r * ldh + k]; mx = fmaxf(mx, v[k]); }
    float s = 0.f;
    float e[10];
    #pragma unroll
    for (int k = 0; k < 10; ++k) { e[k] = __expf(v[k] - mx); s += e[k]; }
    float inv = 1.0f / s;
    #pragma unroll
    for (int k = 0; k < 10; ++k) {
        out_pred[(size_t)r * 10 + k] = e[k] * inv;
        out_h  [(size_t)r * 10 + k] = v[k];
    }
}

// mode 0: identity, 1: clip(exp,1e-5,1e6), 2: clip(softplus,1e-4,1e4), 3: sigmoid
__global__ void sdcn_epilogue(const float* __restrict__ Raw, float* __restrict__ Out,
                              int N, int Nc, int ldr, int mode)
{
    size_t idx = (size_t)blockIdx.x * blockDim.x + threadIdx.x;
    size_t total = (size_t)N * Nc;
    if (idx >= total) return;
    int r = (int)(idx / Nc);
    int j = (int)(idx % Nc);
    float x = Raw[(size_t)r * ldr + j];
    float y;
    if (mode == 0)      y = x;
    else if (mode == 1) y = fminf(fmaxf(__expf(x), 1e-5f), 1e6f);
    else if (mode == 2) { float sp = fmaxf(x, 0.f) + log1pf(__expf(-fabsf(x))); y = fminf(fmaxf(sp, 1e-4f), 1e4f); }
    else                y = 1.0f / (1.0f + __expf(-x));
    Out[idx] = y;
}

__global__ void sdcn_q(const float* __restrict__ Z, int ldz,
                       const float* __restrict__ cluster, float* __restrict__ q, int N)
{
    int r = blockIdx.x * blockDim.x + threadIdx.x;
    if (r >= N) return;
    float zv[10];
    #pragma unroll
    for (int k = 0; k < 10; ++k) zv[k] = Z[(size_t)r * ldz + k];
    float qq[10];
    float s = 0.f;
    #pragma unroll
    for (int c = 0; c < 10; ++c) {
        float d2 = 0.f;
        #pragma unroll
        for (int k = 0; k < 10; ++k) { float t = zv[k] - cluster[c * 10 + k]; d2 += t * t; }
        float val = 1.0f / (1.0f + d2);     // V=1 => exponent (V+1)/2 = 1
        qq[c] = val; s += val;
    }
    float inv = 1.0f / s;
    #pragma unroll
    for (int c = 0; c < 10; ++c) q[(size_t)r * 10 + c] = qq[c] * inv;
}

// ---------------------------------------------------------------------------
// Host orchestration
// ---------------------------------------------------------------------------
static inline void* wsalloc(char* base, size_t& off, size_t bytes) {
    off = (off + 255) & ~(size_t)255;
    void* p = base + off;
    off += bytes;
    return p;
}
static inline unsigned nblk(size_t total, int bs) { return (unsigned)((total + bs - 1) / bs); }

extern "C" void kernel_launch(void* const* d_in, const int* in_sizes, int n_in,
                              void* d_out, int out_size, void* d_ws, size_t ws_size,
                              hipStream_t stream)
{
    (void)n_in; (void)out_size; (void)ws_size;
    const int N = 10000, NIN = 3000, E = in_sizes[1];
    const int Mp = 10048;                 // 10000 -> mult of 64
    const int P500 = 512, P2000 = 2048, P3000 = 3008, P10 = 64;
    const float SIGMA = 0.5f;

    const float* x        = (const float*)d_in[0];
    const int*   esrc     = (const int*)d_in[1];
    const int*   edst     = (const int*)d_in[2];
    const float* ew       = (const float*)d_in[3];
    const float* cluster  = (const float*)d_in[48];

    char* ws = (char*)d_ws;
    size_t off = 0;

    // ---- weight prep: W[K,Nt] f32 -> Wt[Np,Kp] bf16 (transposed, padded) ----
    auto prepW = [&](int widx, int K, int Nt, int Kp, int Np) -> bhalf* {
        bhalf* p = (bhalf*)wsalloc(ws, off, (size_t)Np * Kp * sizeof(bhalf));
        size_t total = (size_t)Np * Kp;
        sdcn_transpose_cast<<<nblk(total, 256), 256, 0, stream>>>(
            (const float*)d_in[widx], p, K, Nt, Kp, Np);
        return p;
    };

    bhalf* Wt_enc1 = prepW( 4, NIN,  500, P3000, P500);
    bhalf* Wt_enc2 = prepW( 8,  500,  500, P500,  P500);
    bhalf* Wt_enc3 = prepW(12,  500, 2000, P500,  P2000);
    bhalf* Wt_z    = prepW(16, 2000,   10, P2000, P10);
    bhalf* Wt_dec1 = prepW(18,   10, 2000, P10,   P2000);
    bhalf* Wt_dec2 = prepW(22, 2000,  500, P2000, P500);
    bhalf* Wt_dec3 = prepW(26,  500,  500, P500,  P500);
    bhalf* Wt_xbar = prepW(30,  500, NIN,  P500,  P3000);
    bhalf* Wt_g1   = prepW(32, NIN,  500, P3000, P500);
    bhalf* Wt_g2   = prepW(34,  500,  500, P500,  P500);
    bhalf* Wt_g3   = prepW(36,  500, 2000, P500,  P2000);
    bhalf* Wt_g4   = prepW(38, 2000,   10, P2000, P10);
    bhalf* Wt_g5   = prepW(40,   10,   10, P10,   P10);
    bhalf* Wt_mean = prepW(42,  500, NIN,  P500,  P3000);
    bhalf* Wt_disp = prepW(44,  500, NIN,  P500,  P3000);
    bhalf* Wt_pi   = prepW(46,  500, NIN,  P500,  P3000);

    // ---- activation buffers ----
    auto af32 = [&](int cols) { return (float*)wsalloc(ws, off, (size_t)Mp * cols * sizeof(float)); };
    auto abf  = [&](int cols) { return (bhalf*)wsalloc(ws, off, (size_t)Mp * cols * sizeof(bhalf)); };

    bhalf* XBF  = abf(P3000);
    float* H1F  = af32(P500);  bhalf* H1B = abf(P500);
    float* H2F  = af32(P500);  bhalf* H2B = abf(P500);
    float* H3F  = af32(P2000); bhalf* H3B = abf(P2000);
    float* ZF   = af32(P10);   bhalf* ZB  = abf(P10);
    float* SUP  = af32(P2000);          // GCN support / decoder d1 raw
    float* GOUT = af32(P2000);          // GCN aggregate / decoder d2 raw
    bhalf* CMB  = abf(P2000);           // GCN combined bf16 / decoder d1 bf16
    float* HG4  = af32(P10);
    float* HG5  = af32(P10);
    bhalf* D2B  = abf(P500);
    float* D3F  = af32(P500);  bhalf* D3B = abf(P500);
    float* BIG  = af32(P3000);          // raw xbar/mean/disp/pi
    float* statM = (float*)wsalloc(ws, off, P2000 * sizeof(float));
    float* statR = (float*)wsalloc(ws, off, P2000 * sizeof(float));

    // ---- helpers ----
    auto gemm = [&](const bhalf* A, const bhalf* Bt, const float* bias, float* C,
                    int Kp, int Np, int Nreal) {
        dim3 grid(Np / 64, Mp / 64);
        sdcn_wmma_gemm_bf16<<<grid, 128, 0, stream>>>(A, Bt, bias, C, Mp, Kp, Np, Nreal);
    };
    auto bn_relu = [&](float* C, bhalf* Cb, int gidx, int bidx, int Nreal, int Np) {
        sdcn_colstats<<<Nreal, 256, 0, stream>>>(C, N, Np, statM, statR);
        size_t total = (size_t)Mp * Np;
        sdcn_bn_relu<<<nblk(total, 256), 256, 0, stream>>>(
            C, Cb, statM, statR, (const float*)d_in[gidx], (const float*)d_in[bidx],
            N, Nreal, Mp, Np);
    };
    auto gcn_agg = [&](const float* sup, float* out, int dout, int ldc, int bidx) {
        size_t total = (size_t)Mp * ldc;
        sdcn_zero<<<nblk(total, 256), 256, 0, stream>>>(out, total);
        dim3 sgrid((unsigned)E, nblk((size_t)dout, 256));
        sdcn_scatter<<<sgrid, 256, 0, stream>>>(sup, out, esrc, edst, ew, dout, ldc);
        size_t btot = (size_t)N * dout;
        sdcn_bias_add<<<nblk(btot, 256), 256, 0, stream>>>(out, (const float*)d_in[bidx], N, dout, ldc);
    };
    auto combine = [&](const float* G, const float* H, bhalf* D, int Nreal, int Np) {
        size_t total = (size_t)Mp * Np;
        sdcn_combine_cast<<<nblk(total, 256), 256, 0, stream>>>(G, H, D, SIGMA, N, Nreal, Mp, Np);
    };

    // ---- output sections ----
    float* out = (float*)d_out;
    float* o_xbar = out;
    float* o_q    = o_xbar + (size_t)N * 3000;
    float* o_pred = o_q    + (size_t)N * 10;
    float* o_h    = o_pred + (size_t)N * 10;
    float* o_mean = o_h    + (size_t)N * 10;
    float* o_disp = o_mean + (size_t)N * 3000;
    float* o_pi   = o_disp + (size_t)N * 3000;

    // ---- input cast ----
    sdcn_cast_pad<<<nblk((size_t)Mp * P3000, 256), 256, 0, stream>>>(x, XBF, N, NIN, Mp, P3000);

    // ---- AE encoder ----
    gemm(XBF, Wt_enc1, (const float*)d_in[5],  H1F, P3000, P500,  500);
    bn_relu(H1F, H1B, 6, 7, 500, P500);
    gemm(H1B, Wt_enc2, (const float*)d_in[9],  H2F, P500,  P500,  500);
    bn_relu(H2F, H2B, 10, 11, 500, P500);
    gemm(H2B, Wt_enc3, (const float*)d_in[13], H3F, P500,  P2000, 2000);
    bn_relu(H3F, H3B, 14, 15, 2000, P2000);
    gemm(H3B, Wt_z,    (const float*)d_in[17], ZF,  P2000, P10,   10);   // no BN

    // ---- GCN stack (sparse aggregation via f32 atomics) ----
    gemm(XBF, Wt_g1, nullptr, SUP, P3000, P500, 500);
    gcn_agg(SUP, GOUT, 500, P500, 33);
    combine(GOUT, H1F, CMB, 500, P500);

    gemm(CMB, Wt_g2, nullptr, SUP, P500, P500, 500);
    gcn_agg(SUP, GOUT, 500, P500, 35);
    combine(GOUT, H2F, CMB, 500, P500);

    gemm(CMB, Wt_g3, nullptr, SUP, P500, P2000, 2000);
    gcn_agg(SUP, GOUT, 2000, P2000, 37);
    combine(GOUT, H3F, CMB, 2000, P2000);

    gemm(CMB, Wt_g4, nullptr, SUP, P2000, P10, 10);
    gcn_agg(SUP, HG4, 10, P10, 39);
    combine(HG4, ZF, CMB, 10, P10);

    gemm(CMB, Wt_g5, nullptr, SUP, P10, P10, 10);
    gcn_agg(SUP, HG5, 10, P10, 41);
    sdcn_softmax_h<<<nblk((size_t)N, 256), 256, 0, stream>>>(HG5, N, P10, o_h, o_pred);

    // ---- AE decoder (reuses GCN slabs) ----
    sdcn_cast_guard<<<nblk((size_t)Mp * P10, 256), 256, 0, stream>>>(ZF, ZB, N, 10, Mp, P10);
    gemm(ZB,  Wt_dec1, (const float*)d_in[19], SUP,  P10,   P2000, 2000);
    bn_relu(SUP, CMB, 20, 21, 2000, P2000);
    gemm(CMB, Wt_dec2, (const float*)d_in[23], GOUT, P2000, P500,  500);
    bn_relu(GOUT, D2B, 24, 25, 500, P500);
    gemm(D2B, Wt_dec3, (const float*)d_in[27], D3F,  P500,  P500,  500);
    bn_relu(D3F, D3B, 28, 29, 500, P500);

    // ---- x_bar + ZINB heads ----
    size_t eptot = (size_t)N * 3000;
    gemm(D3B, Wt_xbar, (const float*)d_in[31], BIG, P500, P3000, 3000);
    sdcn_epilogue<<<nblk(eptot, 256), 256, 0, stream>>>(BIG, o_xbar, N, 3000, P3000, 0);
    gemm(D3B, Wt_mean, (const float*)d_in[43], BIG, P500, P3000, 3000);
    sdcn_epilogue<<<nblk(eptot, 256), 256, 0, stream>>>(BIG, o_mean, N, 3000, P3000, 1);
    gemm(D3B, Wt_disp, (const float*)d_in[45], BIG, P500, P3000, 3000);
    sdcn_epilogue<<<nblk(eptot, 256), 256, 0, stream>>>(BIG, o_disp, N, 3000, P3000, 2);
    gemm(D3B, Wt_pi,   (const float*)d_in[47], BIG, P500, P3000, 3000);
    sdcn_epilogue<<<nblk(eptot, 256), 256, 0, stream>>>(BIG, o_pi, N, 3000, P3000, 3);

    // ---- Student-t q ----
    sdcn_q<<<nblk((size_t)N, 256), 256, 0, stream>>>(ZF, P10, cluster, o_q, N);
}
